// SceneFlowEstimatorProbPointConv2_9354438770935
// MI455X (gfx1250) — matrix operand order repaired and checked by reading, not compiled, not run on hardware
//
#include <hip/hip_runtime.h>

#define BB 2
#define NN 8192
#define KNNK 9

typedef __attribute__((ext_vector_type(16))) _Float16 v16h;
typedef __attribute__((ext_vector_type(8)))  float    v8f;

// ---------------------------------------------------------------------------
// WMMA helper (CDNA5 gfx1250, wave32, V_WMMA_F32_16X16X32_F16)
// ---------------------------------------------------------------------------
__device__ __forceinline__ v8f wmma16(v16h a, v16h b, v8f c) {
  return __builtin_amdgcn_wmma_f32_16x16x32_f16(false, a, false, b, (short)0, c,
                                                false, false);
}

// Fragment-native layouts: one 512-half block per (tile pair); each lane's 16
// halves are contiguous (32B) so fragment load = 2 x b128, no packing movs.
//
// A (16x32 f16): lane l -> m = m0+(l&15); half h -> k = k0+((l&16)?16:0)+h
// B (32x16 f16): lane l -> c = c0+(l&15), k-group = k0+((l&16)?16:0); half h -> k+h

// LDS half-element offset of B element (k, c), ncolt = #column tiles.
__device__ __forceinline__ int bfrag_off(int k, int c, int ncolt) {
  return (((k >> 5) * ncolt + (c >> 4)) * 32 + (c & 15) + (k & 16)) * 16 + (k & 15);
}

__device__ __forceinline__ v16h load_b_frag(const _Float16* Bsh, int ncolt,
                                            int kt, int ct, int lane) {
  return *(const v16h*)(Bsh + (((kt * ncolt) + ct) * 32 + lane) * 16);
}

__device__ __forceinline__ v16h load_a_frag(const _Float16* __restrict__ Wp,
                                            int nkt, int mt, int kt, int lane) {
  return *(const v16h*)(Wp + (((mt * nkt) + kt) * 32 + lane) * 16);
}

__device__ __forceinline__ float lrelu01(float v) { return v > 0.f ? v : 0.1f * v; }

// ---------------------------------------------------------------------------
// Kernel 0: pack an f32 weight matrix W[M][kvalid] into f16 A-fragment layout
// (kpad = kvalid rounded up to 32; zero-padded).
// ---------------------------------------------------------------------------
__global__ void pack_weight_kernel(const float* __restrict__ W,
                                   _Float16* __restrict__ out,
                                   int M, int kvalid, int kpad) {
  int total = M * kpad;
  int nkt = kpad >> 5;
  for (int t = blockIdx.x * blockDim.x + threadIdx.x; t < total;
       t += gridDim.x * blockDim.x) {
    int frag = t >> 9;            // (mt, kt)
    int within = t & 511;         // lane*16 + h
    int lane = within >> 4, h = within & 15;
    int mt = frag / nkt, kt = frag - mt * nkt;
    int m = mt * 16 + (lane & 15);
    int k = kt * 32 + ((lane & 16) ? 16 : 0) + h;
    float v = (k < kvalid) ? W[m * kvalid + k] : 0.f;
    out[t] = (_Float16)v;
  }
}

// ---------------------------------------------------------------------------
// Kernel 1: brute-force self-KNN (top-9 nearest incl. self, ascending dist)
// ---------------------------------------------------------------------------
__global__ void knn_kernel(const float* __restrict__ xyz, int* __restrict__ idx_out) {
  const int b = blockIdx.y;
  const int n = blockIdx.x * blockDim.x + threadIdx.x;
  const float* X = xyz + (size_t)b * 3 * NN;
  const float qx = X[n], qy = X[NN + n], qz = X[2 * NN + n];

  float bd[KNNK];
  int   bi[KNNK];
#pragma unroll
  for (int k = 0; k < KNNK; ++k) { bd[k] = 3.4e38f; bi[k] = 0; }

  __shared__ float sx[256], sy[256], sz[256];
  for (int t = 0; t < NN; t += 256) {
    sx[threadIdx.x] = X[t + threadIdx.x];
    sy[threadIdx.x] = X[NN + t + threadIdx.x];
    sz[threadIdx.x] = X[2 * NN + t + threadIdx.x];
    __syncthreads();
    for (int j = 0; j < 256; ++j) {
      float dx = qx - sx[j], dy = qy - sy[j], dz = qz - sz[j];
      float d = dx * dx + dy * dy + dz * dz;
      if (d < bd[KNNK - 1]) {
        bd[KNNK - 1] = d; bi[KNNK - 1] = t + j;
#pragma unroll
        for (int jj = KNNK - 1; jj > 0; --jj) {
          if (bd[jj] < bd[jj - 1]) {
            float td = bd[jj]; bd[jj] = bd[jj - 1]; bd[jj - 1] = td;
            int   ti = bi[jj]; bi[jj] = bi[jj - 1]; bi[jj - 1] = ti;
          }
        }
      }
    }
    __syncthreads();
  }
#pragma unroll
  for (int k = 0; k < KNNK; ++k)
    idx_out[((size_t)b * NN + n) * KNNK + k] = bi[k];
}

// ---------------------------------------------------------------------------
// Kernel 2: gather + conv0(67->128) + conv1(128->128) (BN+LeakyReLU) + max_k
// One workgroup = 8 points -> 72 cols padded to 80 (5 col tiles).
// ---------------------------------------------------------------------------
__global__ void conv_knn_kernel(const float* __restrict__ cv,
                                const float* __restrict__ flow,
                                const int* __restrict__ idx_ws,
                                const _Float16* __restrict__ w0p,  // nkt=3
                                const float* __restrict__ b0,
                                const float* __restrict__ bn0,
                                const _Float16* __restrict__ w1p,  // nkt=4
                                const float* __restrict__ b1,
                                const float* __restrict__ bn1,
                                float* __restrict__ h_out) {
  __shared__ int idx_sh[72];
  __shared__ __align__(32) _Float16 Gsh[3 * 5 * 512];   // B-frag, k:96, c:80
  __shared__ __align__(32) _Float16 H1sh[4 * 5 * 512];  // B-frag, k:128, c:80
  __shared__ _Float16 H2sh[128 * 80];                   // row-major for max

  const int tid = threadIdx.x;
  const int b = blockIdx.y;
  const int n0 = blockIdx.x * 8;

  if (tid < 72) idx_sh[tid] = idx_ws[((size_t)b * NN + n0) * KNNK + tid];
  __syncthreads();

  for (int t = tid; t < 96 * 80; t += 256) {
    int f = t / 80, c = t - (t / 80) * 80;
    float v = 0.f;
    if (f < 67 && c < 72) {
      int m = idx_sh[c];
      v = (f < 64) ? cv[((size_t)b * 64 + f) * NN + m]
                   : flow[((size_t)b * 3 + (f - 64)) * NN + m];
    }
    Gsh[bfrag_off(f, c, 5)] = (_Float16)v;
  }
  __syncthreads();

  const int lane = tid & 31, wave = tid >> 5;
  const int m0 = wave * 16;
  const int hi8 = (lane & 16) ? 8 : 0;
  const int cl = lane & 15;

  // ---- GEMM1: W0[128,96p] x G[96,80] ----
  {
    float s0[8], t0[8];
#pragma unroll
    for (int r = 0; r < 8; ++r) {
      int m = m0 + r + hi8;
      float s = bn0[m] * rsqrtf(bn0[384 + m] + 1e-5f);
      s0[r] = s;
      t0[r] = bn0[128 + m] + (b0[m] - bn0[256 + m]) * s;
    }
    v16h a0 = load_a_frag(w0p, 3, wave, 0, lane);
    v16h a1 = load_a_frag(w0p, 3, wave, 1, lane);
    v16h a2 = load_a_frag(w0p, 3, wave, 2, lane);
    for (int ct = 0; ct < 5; ++ct) {
      v8f acc = {};
      acc = wmma16(a0, load_b_frag(Gsh, 5, 0, ct, lane), acc);
      acc = wmma16(a1, load_b_frag(Gsh, 5, 1, ct, lane), acc);
      acc = wmma16(a2, load_b_frag(Gsh, 5, 2, ct, lane), acc);
      int c = 16 * ct + cl;
#pragma unroll
      for (int r = 0; r < 8; ++r) {
        int m = m0 + r + hi8;
        H1sh[bfrag_off(m, c, 5)] = (_Float16)lrelu01(acc[r] * s0[r] + t0[r]);
      }
    }
  }
  __syncthreads();

  // ---- GEMM2: W1[128,128] x H1[128,80] ----
  {
    float s1[8], t1[8];
#pragma unroll
    for (int r = 0; r < 8; ++r) {
      int m = m0 + r + hi8;
      float s = bn1[m] * rsqrtf(bn1[384 + m] + 1e-5f);
      s1[r] = s;
      t1[r] = bn1[128 + m] + (b1[m] - bn1[256 + m]) * s;
    }
    v16h a0 = load_a_frag(w1p, 4, wave, 0, lane);
    v16h a1 = load_a_frag(w1p, 4, wave, 1, lane);
    v16h a2 = load_a_frag(w1p, 4, wave, 2, lane);
    v16h a3 = load_a_frag(w1p, 4, wave, 3, lane);
    for (int ct = 0; ct < 5; ++ct) {
      v8f acc = {};
      acc = wmma16(a0, load_b_frag(H1sh, 5, 0, ct, lane), acc);
      acc = wmma16(a1, load_b_frag(H1sh, 5, 1, ct, lane), acc);
      acc = wmma16(a2, load_b_frag(H1sh, 5, 2, ct, lane), acc);
      acc = wmma16(a3, load_b_frag(H1sh, 5, 3, ct, lane), acc);
      int c = 16 * ct + cl;
#pragma unroll
      for (int r = 0; r < 8; ++r) {
        int m = m0 + r + hi8;
        H2sh[m * 80 + c] = (_Float16)lrelu01(acc[r] * s1[r] + t1[r]);
      }
    }
  }
  __syncthreads();

  // ---- max over k=0..8 ----
  for (int t = tid; t < 128 * 8; t += 256) {
    int ch = t >> 3, p = t & 7;
    float mx = -3.4e38f;
#pragma unroll
    for (int k = 0; k < KNNK; ++k)
      mx = fmaxf(mx, (float)H2sh[ch * 80 + p * KNNK + k]);
    h_out[((size_t)b * 128 + ch) * NN + n0 + p] = mx;
  }
}

// ---------------------------------------------------------------------------
// Kernel 3: ConvGRU. One workgroup = 16 points (1 col tile); 8 waves = M 128.
// ---------------------------------------------------------------------------
__global__ void gru_kernel(const float* __restrict__ h_ws,
                           const float* __restrict__ feats,
                           const _Float16* __restrict__ wzp,  // nkt=8
                           const float* __restrict__ bz,
                           const _Float16* __restrict__ wrp,
                           const float* __restrict__ br,
                           const _Float16* __restrict__ wqp,
                           const float* __restrict__ bq,
                           float* __restrict__ hn_ws) {
  __shared__ __align__(32) _Float16 hxh[8 * 512];  // B-frag, k:256 (h|feats), c:16
  __shared__ float hf[128 * 16];

  const int tid = threadIdx.x;
  const int b = blockIdx.y;
  const int n0 = blockIdx.x * 16;

  for (int t = tid; t < 256 * 16; t += 256) {
    int row = t >> 4, c = t & 15, n = n0 + c;
    float v = (row < 128) ? h_ws[((size_t)b * 128 + row) * NN + n]
                          : feats[((size_t)b * 128 + (row - 128)) * NN + n];
    hxh[bfrag_off(row, c, 1)] = (_Float16)v;
    if (row < 128) hf[row * 16 + c] = v;
  }
  __syncthreads();

  const int lane = tid & 31, wave = tid >> 5;
  const int m0 = wave * 16;
  const int hi8 = (lane & 16) ? 8 : 0;
  const int cl = lane & 15;

  v8f az = {}, ar = {};
#pragma unroll
  for (int ks = 0; ks < 8; ++ks)
    az = wmma16(load_a_frag(wzp, 8, wave, ks, lane),
                load_b_frag(hxh, 1, ks, 0, lane), az);
#pragma unroll
  for (int ks = 0; ks < 8; ++ks)
    ar = wmma16(load_a_frag(wrp, 8, wave, ks, lane),
                load_b_frag(hxh, 1, ks, 0, lane), ar);

  float zz[8], rr[8];
#pragma unroll
  for (int r = 0; r < 8; ++r) {
    int m = m0 + r + hi8;
    zz[r] = 1.f / (1.f + expf(-(az[r] + bz[m])));
    rr[r] = 1.f / (1.f + expf(-(ar[r] + br[m])));
  }
  __syncthreads();  // all gate reads of hxh done
#pragma unroll
  for (int r = 0; r < 8; ++r) {
    int m = m0 + r + hi8;
    hxh[bfrag_off(m, cl, 1)] = (_Float16)(rr[r] * hf[m * 16 + cl]);  // r*h
  }
  __syncthreads();

  v8f aq = {};
#pragma unroll
  for (int ks = 0; ks < 8; ++ks)
    aq = wmma16(load_a_frag(wqp, 8, wave, ks, lane),
                load_b_frag(hxh, 1, ks, 0, lane), aq);
#pragma unroll
  for (int r = 0; r < 8; ++r) {
    int m = m0 + r + hi8;
    float q = tanhf(aq[r] + bq[m]);
    float hv = hf[m * 16 + cl];
    hn_ws[((size_t)b * 128 + m) * NN + n0 + cl] = (1.f - zz[r]) * hv + zz[r] * q;
  }
}

// ---------------------------------------------------------------------------
// Kernel 4: MLP head + refine/coarse regnets + final flow combine.
// ---------------------------------------------------------------------------
__global__ void head_kernel(const float* __restrict__ points,
                            const float* __restrict__ hn_ws,
                            const float* __restrict__ cv,
                            const float* __restrict__ flow,
                            const _Float16* __restrict__ w2ap,  // nkt=6
                            const float* __restrict__ b2a,
                            const _Float16* __restrict__ w2bp,  // nkt=4
                            const float* __restrict__ b2b,
                            const float* __restrict__ wfc, const float* __restrict__ bfc,
                            const _Float16* __restrict__ rw0p, const float* __restrict__ rbn0,
                            const _Float16* __restrict__ rw1p, const float* __restrict__ rbn1,
                            const float* __restrict__ rw2, const float* __restrict__ rbn2,
                            const _Float16* __restrict__ cw0p, const float* __restrict__ cbn0,
                            const _Float16* __restrict__ cw1p, const float* __restrict__ cbn1,
                            const float* __restrict__ cw2, const float* __restrict__ cbn2,
                            float* __restrict__ out_x, float* __restrict__ out_flow) {
  __shared__ __align__(32) _Float16 in0h[6 * 512];  // k:192 (points|h), c:16
  __shared__ __align__(32) _Float16 x1h[4 * 512];   // k:128
  __shared__ float xf[64 * 16];
  __shared__ __align__(32) _Float16 xh[2 * 512];    // k:64
  __shared__ __align__(32) _Float16 cvh[2 * 512];   // k:64
  __shared__ __align__(32) _Float16 ya[2][2 * 512];
  __shared__ __align__(32) _Float16 yb[2][2 * 512];
  __shared__ float rfl[3 * 16];
  __shared__ float rprob[16], cprob[16];

  const int tid = threadIdx.x;
  const int b = blockIdx.y;
  const int n0 = blockIdx.x * 16;

  for (int t = tid; t < 192 * 16; t += 256) {
    int row = t >> 4, c = t & 15, n = n0 + c;
    float v = (row < 64) ? points[((size_t)b * 64 + row) * NN + n]
                         : hn_ws[((size_t)b * 128 + (row - 64)) * NN + n];
    in0h[bfrag_off(row, c, 1)] = (_Float16)v;
  }
  for (int t = tid; t < 64 * 16; t += 256) {
    int row = t >> 4, c = t & 15;
    cvh[bfrag_off(row, c, 1)] = (_Float16)cv[((size_t)b * 64 + row) * NN + n0 + c];
  }
  __syncthreads();

  const int lane = tid & 31, wave = tid >> 5;
  const int hi8 = (lane & 16) ? 8 : 0;
  const int cl = lane & 15;

  {  // w2a: [128,192]
    int m0 = wave * 16;
    v8f acc = {};
#pragma unroll
    for (int ks = 0; ks < 6; ++ks)
      acc = wmma16(load_a_frag(w2ap, 6, wave, ks, lane),
                   load_b_frag(in0h, 1, ks, 0, lane), acc);
#pragma unroll
    for (int r = 0; r < 8; ++r) {
      int m = m0 + r + hi8;
      x1h[bfrag_off(m, cl, 1)] = (_Float16)lrelu01(acc[r] + b2a[m]);
    }
  }
  __syncthreads();

  if (wave < 4) {  // w2b: [64,128]; writes x to d_out
    int m0 = wave * 16;
    v8f acc = {};
#pragma unroll
    for (int ks = 0; ks < 4; ++ks)
      acc = wmma16(load_a_frag(w2bp, 4, wave, ks, lane),
                   load_b_frag(x1h, 1, ks, 0, lane), acc);
#pragma unroll
    for (int r = 0; r < 8; ++r) {
      int m = m0 + r + hi8;
      float v = lrelu01(acc[r] + b2b[m]);
      xf[m * 16 + cl] = v;
      xh[bfrag_off(m, cl, 1)] = (_Float16)v;
      out_x[((size_t)b * 64 + m) * NN + n0 + cl] = v;
    }
  }
  __syncthreads();

  if (tid < 48) {  // wfc: [3,64] scalar
    int o = tid >> 4, c = tid & 15;
    float s = bfc[o];
    for (int j = 0; j < 64; ++j) s += wfc[o * 64 + j] * xf[j * 16 + c];
    rfl[o * 16 + c] = s;
  }

  {  // regnet layer 0 (refine: waves 0-3 on x; coarse: waves 4-7 on cv)
    const _Float16* Wp; const float* BN; const _Float16* IN; _Float16* OUT;
    int mt = wave & 3;
    if (wave < 4) { Wp = rw0p; BN = rbn0; IN = xh;  OUT = &ya[0][0]; }
    else          { Wp = cw0p; BN = cbn0; IN = cvh; OUT = &ya[1][0]; }
    int m0 = mt * 16;
    v8f acc = {};
#pragma unroll
    for (int ks = 0; ks < 2; ++ks)
      acc = wmma16(load_a_frag(Wp, 2, mt, ks, lane),
                   load_b_frag(IN, 1, ks, 0, lane), acc);
#pragma unroll
    for (int r = 0; r < 8; ++r) {
      int m = m0 + r + hi8;
      float s = BN[m] * rsqrtf(BN[192 + m] + 1e-5f);
      float t = BN[64 + m] - BN[128 + m] * s;
      float v = acc[r] * s + t;
      OUT[bfrag_off(m, cl, 1)] = (_Float16)(v > 0.f ? v : 0.f);
    }
  }
  __syncthreads();

  {  // regnet layer 1
    const _Float16* Wp; const float* BN; const _Float16* IN; _Float16* OUT;
    int mt = wave & 3;
    if (wave < 4) { Wp = rw1p; BN = rbn1; IN = &ya[0][0]; OUT = &yb[0][0]; }
    else          { Wp = cw1p; BN = cbn1; IN = &ya[1][0]; OUT = &yb[1][0]; }
    int m0 = mt * 16;
    v8f acc = {};
#pragma unroll
    for (int ks = 0; ks < 2; ++ks)
      acc = wmma16(load_a_frag(Wp, 2, mt, ks, lane),
                   load_b_frag(IN, 1, ks, 0, lane), acc);
#pragma unroll
    for (int r = 0; r < 8; ++r) {
      int m = m0 + r + hi8;
      float s = BN[m] * rsqrtf(BN[192 + m] + 1e-5f);
      float t = BN[64 + m] - BN[128 + m] * s;
      float v = acc[r] * s + t;
      OUT[bfrag_off(m, cl, 1)] = (_Float16)(v > 0.f ? v : 0.f);
    }
  }
  __syncthreads();

  if (tid < 16) {  // refine final [1,64] + BN(C=1) + relu
    float s = 0.f;
    for (int j = 0; j < 64; ++j) s += rw2[j] * (float)yb[0][bfrag_off(j, tid, 1)];
    float sc = rbn2[0] * rsqrtf(rbn2[3] + 1e-5f);
    float v = s * sc + (rbn2[1] - rbn2[2] * sc);
    rprob[tid] = v > 0.f ? v : 0.f;
  }
  if (tid >= 128 && tid < 144) {  // coarse final
    int c = tid - 128;
    float s = 0.f;
    for (int j = 0; j < 64; ++j) s += cw2[j] * (float)yb[1][bfrag_off(j, c, 1)];
    float sc = cbn2[0] * rsqrtf(cbn2[3] + 1e-5f);
    float v = s * sc + (cbn2[1] - cbn2[2] * sc);
    cprob[c] = v > 0.f ? v : 0.f;
  }
  __syncthreads();

  if (tid < 48) {  // re_flow = clip(coarse*flow + refine*re_flow)
    int o = tid >> 4, c = tid & 15, n = n0 + c;
    float fl = flow[((size_t)b * 3 + o) * NN + n];
    float v = cprob[c] * fl + rprob[c] * rfl[o * 16 + c];
    v = fminf(20.f, fmaxf(-20.f, v));
    out_flow[((size_t)b * 3 + o) * NN + n] = v;
  }
}

// ---------------------------------------------------------------------------
extern "C" void kernel_launch(void* const* d_in, const int* in_sizes, int n_in,
                              void* d_out, int out_size, void* d_ws, size_t ws_size,
                              hipStream_t stream) {
  const float* xyz    = (const float*)d_in[0];
  const float* points = (const float*)d_in[1];
  const float* cv     = (const float*)d_in[2];
  const float* feats  = (const float*)d_in[3];
  const float* flow   = (const float*)d_in[4];
  const float* w0  = (const float*)d_in[5];
  const float* b0  = (const float*)d_in[6];
  const float* bn0 = (const float*)d_in[7];
  const float* w1  = (const float*)d_in[8];
  const float* b1  = (const float*)d_in[9];
  const float* bn1 = (const float*)d_in[10];
  const float* wz = (const float*)d_in[11];
  const float* bz = (const float*)d_in[12];
  const float* wr = (const float*)d_in[13];
  const float* br = (const float*)d_in[14];
  const float* wq = (const float*)d_in[15];
  const float* bq = (const float*)d_in[16];
  const float* w2a = (const float*)d_in[17];
  const float* b2a = (const float*)d_in[18];
  const float* w2b = (const float*)d_in[19];
  const float* b2b = (const float*)d_in[20];
  const float* wfc = (const float*)d_in[21];
  const float* bfc = (const float*)d_in[22];
  const float* rw0  = (const float*)d_in[23];
  const float* rbn0 = (const float*)d_in[24];
  const float* rw1  = (const float*)d_in[25];
  const float* rbn1 = (const float*)d_in[26];
  const float* rw2  = (const float*)d_in[27];
  const float* rbn2 = (const float*)d_in[28];
  const float* cw0  = (const float*)d_in[29];
  const float* cbn0 = (const float*)d_in[30];
  const float* cw1  = (const float*)d_in[31];
  const float* cbn1 = (const float*)d_in[32];
  const float* cw2  = (const float*)d_in[33];
  const float* cbn2 = (const float*)d_in[34];

  float* out_x = (float*)d_out;                 // [B,64,N]
  float* out_f = out_x + (size_t)BB * 64 * NN;  // [B,3,N]

  // workspace layout
  int*      idx_ws = (int*)d_ws;                                     // @0, 0.57MB
  _Float16* wpk    = (_Float16*)((char*)d_ws + 0xA0000);             // @640KB, 344KB
  float*    h_ws   = (float*)((char*)d_ws + ((size_t)1 << 20));      // @1MB, 8MB
  float*    hn_ws  = (float*)((char*)d_ws + ((size_t)9 << 20));      // @9MB, 8MB

  _Float16* w0p  = wpk;            // 128x96  = 12288
  _Float16* w1p  = wpk + 12288;    // 128x128 = 16384
  _Float16* wzp  = wpk + 28672;    // 128x256 = 32768
  _Float16* wrp  = wpk + 61440;
  _Float16* wqp  = wpk + 94208;
  _Float16* w2ap = wpk + 126976;   // 128x192 = 24576
  _Float16* w2bp = wpk + 151552;   // 64x128  = 8192
  _Float16* rw0p = wpk + 159744;   // 64x64   = 4096
  _Float16* rw1p = wpk + 163840;
  _Float16* cw0p = wpk + 167936;
  _Float16* cw1p = wpk + 172032;

  pack_weight_kernel<<<48, 256, 0, stream>>>(w0, w0p, 128, 67, 96);
  pack_weight_kernel<<<64, 256, 0, stream>>>(w1, w1p, 128, 128, 128);
  pack_weight_kernel<<<128, 256, 0, stream>>>(wz, wzp, 128, 256, 256);
  pack_weight_kernel<<<128, 256, 0, stream>>>(wr, wrp, 128, 256, 256);
  pack_weight_kernel<<<128, 256, 0, stream>>>(wq, wqp, 128, 256, 256);
  pack_weight_kernel<<<96, 256, 0, stream>>>(w2a, w2ap, 128, 192, 192);
  pack_weight_kernel<<<32, 256, 0, stream>>>(w2b, w2bp, 64, 128, 128);
  pack_weight_kernel<<<16, 256, 0, stream>>>(rw0, rw0p, 64, 64, 64);
  pack_weight_kernel<<<16, 256, 0, stream>>>(rw1, rw1p, 64, 64, 64);
  pack_weight_kernel<<<16, 256, 0, stream>>>(cw0, cw0p, 64, 64, 64);
  pack_weight_kernel<<<16, 256, 0, stream>>>(cw1, cw1p, 64, 64, 64);

  knn_kernel<<<dim3(NN / 256, BB), 256, 0, stream>>>(xyz, idx_ws);

  conv_knn_kernel<<<dim3(NN / 8, BB), 256, 0, stream>>>(
      cv, flow, idx_ws, w0p, b0, bn0, w1p, b1, bn1, h_ws);

  gru_kernel<<<dim3(NN / 16, BB), 256, 0, stream>>>(
      h_ws, feats, wzp, bz, wrp, br, wqp, bq, hn_ws);

  head_kernel<<<dim3(NN / 16, BB), 256, 0, stream>>>(
      points, hn_ws, cv, flow, w2ap, b2a, w2bp, b2b, wfc, bfc,
      rw0p, rbn0, rw1p, rbn1, rw2, rbn2,
      cw0p, cbn0, cw1p, cbn1, cw2, cbn2, out_x, out_f);

  (void)in_sizes; (void)n_in; (void)out_size; (void)ws_size;
}
